// MultiHeaderAttention_63874753626331
// MI455X (gfx1250) — compile-verified, matrix operand order checked by reading, not run on hardware
//
#include <hip/hip_runtime.h>

// ---------------------------------------------------------------------------
// Types for CDNA5 WMMA
// ---------------------------------------------------------------------------
typedef __attribute__((ext_vector_type(16))) __bf16 v16bf;
typedef __attribute__((ext_vector_type(8)))  float  v8f;
typedef int v4i __attribute__((vector_size(16)));   // matches builtin param

#define D_MODEL 1024
#define S_LEN   2048
#define N_BATCH 4
#define N_HEADS 16
#define HD      64
#define M_ROWS  (N_BATCH * S_LEN)   // 8192

// CDNA5 async global->LDS path (ASYNCcnt), guarded so compile never breaks.
#if defined(__AMDGCN__) && __has_builtin(__builtin_amdgcn_global_load_async_to_lds_b128) && __has_builtin(__builtin_amdgcn_s_wait_asynccnt)
#define USE_ASYNC 1
#else
#define USE_ASYNC 0
#endif

// ---------------------------------------------------------------------------
// Helpers
// ---------------------------------------------------------------------------
__device__ __forceinline__ unsigned short f32_to_bf16(float f) {
    unsigned int x = __float_as_uint(f);
    x += 0x7FFFu + ((x >> 16) & 1u);   // round to nearest even
    return (unsigned short)(x >> 16);
}

__device__ __forceinline__ unsigned int pack_bf16x2(float a, float b) {
    return ((unsigned int)f32_to_bf16(b) << 16) | (unsigned int)f32_to_bf16(a);
}

__device__ __forceinline__ v8f zero8() {
    v8f z;
#pragma unroll
    for (int i = 0; i < 8; ++i) z[i] = 0.0f;
    return z;
}

#if USE_ASYNC
__device__ __forceinline__ void async_copy16(const unsigned short* g,
                                             unsigned short* l) {
    __builtin_amdgcn_global_load_async_to_lds_b128(
        (__attribute__((address_space(1))) v4i*)g,
        (__attribute__((address_space(3))) v4i*)l,
        /*offset=*/0, /*cpol=*/0);
}
#endif

// Load a 16x32 bf16 fragment (A-layout; B mirrors it with N in place of M)
// from LDS stored row-major with `rowStride` (in ushorts).
// Lane L holds row (rowBase + (L&15)); K pairs per ISA table:
// lanes 0-15: k = {0,2,4,6,16,18,20,22}; lanes 16-31: +8.
__device__ __forceinline__ v16bf load_frag(const unsigned short* base,
                                           int rowStride, int rowBase,
                                           int kOff, int lane) {
    union { v16bf v; unsigned int u[8]; } f;
    const int row = rowBase + (lane & 15);
    const int kb  = kOff + ((lane & 16) ? 8 : 0);
    const unsigned short* p = base + (size_t)row * rowStride;
#pragma unroll
    for (int r = 0; r < 8; ++r) {
        const int k = kb + ((r < 4) ? (2 * r) : (16 + 2 * (r - 4)));
        f.u[r] = *(const unsigned int*)(p + k);
    }
    return f.v;
}

__device__ __forceinline__ v16bf ones_frag() {
    union { v16bf v; unsigned int u[8]; } f;
#pragma unroll
    for (int i = 0; i < 8; ++i) f.u[i] = 0x3F803F80u;   // bf16(1.0) x2
    return f.v;
}

__device__ __forceinline__ v8f wmma_bf16(v16bf a, v16bf b, v8f c) {
    return __builtin_amdgcn_wmma_f32_16x16x32_bf16(
        false, a, false, b, (short)0, c, false, false);
}

// ---------------------------------------------------------------------------
// Kernel 1: weight transpose + f32 -> bf16 convert.  W:[K][N] -> Wt:[N][K]
// ---------------------------------------------------------------------------
__global__ __launch_bounds__(256) void wtrans_kernel(
    const float* __restrict__ W, unsigned short* __restrict__ Wt, int D) {
    __shared__ float tile[32][33];
    const int k0 = blockIdx.x * 32;
    const int n0 = blockIdx.y * 32;
    const int x = threadIdx.x;   // 0..31
    const int y = threadIdx.y;   // 0..7
#pragma unroll
    for (int i = 0; i < 4; ++i)
        tile[y + 8 * i][x] = W[(size_t)(k0 + y + 8 * i) * D + n0 + x];
    __syncthreads();
#pragma unroll
    for (int i = 0; i < 4; ++i)
        Wt[(size_t)(n0 + y + 8 * i) * D + (k0 + x)] =
            f32_to_bf16(tile[x][y + 8 * i]);
}

// ---------------------------------------------------------------------------
// Kernel 2: elementwise f32 -> bf16 (activations), 8 elements/thread.
// ---------------------------------------------------------------------------
__global__ __launch_bounds__(256) void cvt_bf16_kernel(
    const float* __restrict__ src, unsigned short* __restrict__ dst, int n8) {
    const int i = blockIdx.x * 256 + threadIdx.x;
    if (i >= n8) return;
    const float4 a = ((const float4*)src)[2 * i];
    const float4 b = ((const float4*)src)[2 * i + 1];
    uint4 o;
    o.x = pack_bf16x2(a.x, a.y);
    o.y = pack_bf16x2(a.z, a.w);
    o.z = pack_bf16x2(b.x, b.y);
    o.w = pack_bf16x2(b.z, b.w);
    ((uint4*)dst)[i] = o;
}

// ---------------------------------------------------------------------------
// Kernel 3: bf16 GEMM via WMMA:  C[M,N] = A[M,K] @ Bt^T   (Bt: [N][K] bf16)
// Block tile 256x128, 8 waves (4x2), wave tile 64x64 (16 WMMA accumulators),
// K-step 32, double-buffered LDS with async global->LDS copies.
// ---------------------------------------------------------------------------
template <bool OUT_F32>
__global__ __launch_bounds__(256) void gemm_bf16_wmma_kernel(
    const unsigned short* __restrict__ A,
    const unsigned short* __restrict__ Bt,
    void* __restrict__ Cptr, int M, int N, int K) {
    __shared__ unsigned short lA[2][256 * 40];   // 32 k + pad(8)
    __shared__ unsigned short lB[2][128 * 40];

    const int t    = threadIdx.x;
    const int lane = t & 31;
    const int w    = t >> 5;
    const int wm   = w >> 1;   // 0..3 -> 64 rows each
    const int wn   = w & 1;    // 0..1 -> 64 cols each
    const int m0   = blockIdx.y * 256;
    const int n0   = blockIdx.x * 128;

    v8f acc[4][4];
#pragma unroll
    for (int mt = 0; mt < 4; ++mt)
#pragma unroll
        for (int nt = 0; nt < 4; ++nt) acc[mt][nt] = zero8();

    // staging geometry: A row = t (64B each), B row = t>>1 (32B halves)
    const int rA = t;
    const int rB = t >> 1;
    const int hB = t & 1;
    const unsigned short* gA = A  + (size_t)(m0 + rA) * K;
    const unsigned short* gB = Bt + (size_t)(n0 + rB) * K + hB * 16;

    auto stage = [&](int buf, int k0) {
        unsigned short* dA = &lA[buf][rA * 40];
        unsigned short* dB = &lB[buf][rB * 40 + hB * 16];
#if USE_ASYNC
#pragma unroll
        for (int i = 0; i < 4; ++i) async_copy16(gA + k0 + 8 * i, dA + 8 * i);
#pragma unroll
        for (int i = 0; i < 2; ++i) async_copy16(gB + k0 + 8 * i, dB + 8 * i);
#else
#pragma unroll
        for (int i = 0; i < 4; ++i)
            *(uint4*)(dA + 8 * i) = *(const uint4*)(gA + k0 + 8 * i);
#pragma unroll
        for (int i = 0; i < 2; ++i)
            *(uint4*)(dB + 8 * i) = *(const uint4*)(gB + k0 + 8 * i);
#endif
    };

    stage(0, 0);
    const int NK = K / 32;
    for (int kt = 0; kt < NK; ++kt) {
        const int cur = kt & 1;
        if (kt + 1 < NK) {
            stage(cur ^ 1, (kt + 1) * 32);   // overlaps with this step's math
#if USE_ASYNC
            __builtin_amdgcn_s_wait_asynccnt(6);   // 6 just issued; prev done
#endif
        } else {
#if USE_ASYNC
            __builtin_amdgcn_s_wait_asynccnt(0);
#endif
        }
        __syncthreads();

        v16bf aF[4], bF[4];
#pragma unroll
        for (int mt = 0; mt < 4; ++mt)
            aF[mt] = load_frag(&lA[cur][0], 40, wm * 64 + mt * 16, 0, lane);
#pragma unroll
        for (int nt = 0; nt < 4; ++nt)
            bF[nt] = load_frag(&lB[cur][0], 40, wn * 64 + nt * 16, 0, lane);
#pragma unroll
        for (int mt = 0; mt < 4; ++mt)
#pragma unroll
            for (int nt = 0; nt < 4; ++nt)
                acc[mt][nt] = wmma_bf16(aF[mt], bF[nt], acc[mt][nt]);
        __syncthreads();
    }

    // epilogue (C/D layout: VGPR r -> M = r + (lane>=16 ? 8 : 0))
    const int hi = (lane & 16) ? 8 : 0;
#pragma unroll
    for (int mt = 0; mt < 4; ++mt)
#pragma unroll
        for (int nt = 0; nt < 4; ++nt)
#pragma unroll
            for (int r = 0; r < 8; ++r) {
                const int gr = m0 + wm * 64 + mt * 16 + r + hi;
                const int gc = n0 + wn * 64 + nt * 16 + (lane & 15);
                if (OUT_F32)
                    ((float*)Cptr)[(size_t)gr * N + gc] = acc[mt][nt][r];
                else
                    ((unsigned short*)Cptr)[(size_t)gr * N + gc] =
                        f32_to_bf16(acc[mt][nt][r]);
            }
}

// ---------------------------------------------------------------------------
// Kernel 4: flash attention per (b,h) slab (contiguous [S][64] bf16).
// 8 waves x 16 query rows; keys in chunks of 32. Row-sums of P are computed
// with an extra WMMA against an all-ones B matrix (lands in C/D row layout),
// only the row-max needs cross-lane shuffles.
// ---------------------------------------------------------------------------
__global__ __launch_bounds__(256) void attn_kernel(
    const unsigned short* __restrict__ Qg,
    const unsigned short* __restrict__ Kg,
    const unsigned short* __restrict__ Vg,
    unsigned short* __restrict__ Og) {
    __shared__ unsigned short sQ[128 * 72];
    __shared__ unsigned short sK[32 * 72];
    __shared__ unsigned short sVt[64 * 40];
    __shared__ unsigned short sP[8 * 16 * 40];

    const int t    = threadIdx.x;
    const int lane = t & 31;
    const int w    = t >> 5;
    const int q0   = blockIdx.x * 128;
    const size_t base = (size_t)blockIdx.y * S_LEN * HD;

    {   // stage Q block 128x64
        const int r = t >> 1, hq = t & 1;
        const unsigned short* src = Qg + base + (size_t)(q0 + r) * HD + hq * 32;
        unsigned short* dst = sQ + r * 72 + hq * 32;
#pragma unroll
        for (int j = 0; j < 4; ++j)
            *(uint4*)(dst + j * 8) = *(const uint4*)(src + j * 8);
    }
    __syncthreads();
    v16bf qF[2];
    qF[0] = load_frag(sQ, 72, w * 16, 0, lane);
    qF[1] = load_frag(sQ, 72, w * 16, 32, lane);
    const v16bf oneF = ones_frag();

    float m_[8], l_[8];
    v8f o[4];
#pragma unroll
    for (int r = 0; r < 8; ++r) { m_[r] = -1e30f; l_[r] = 0.0f; }
#pragma unroll
    for (int nt = 0; nt < 4; ++nt) o[nt] = zero8();

    const int hi = (lane & 16) ? 8 : 0;

    for (int c0 = 0; c0 < S_LEN; c0 += 32) {
        __syncthreads();
        if (t < 128) {   // K chunk row-major
            const int r = t >> 2, i = t & 3;
            const unsigned short* src = Kg + base + (size_t)(c0 + r) * HD + i * 16;
            unsigned short* dst = sK + r * 72 + i * 16;
            *(uint4*)(dst)     = *(const uint4*)(src);
            *(uint4*)(dst + 8) = *(const uint4*)(src + 8);
        } else {         // V chunk transposed
            const int u = t - 128;
            const int r = u >> 2, i = u & 3;
            const unsigned short* src = Vg + base + (size_t)(c0 + r) * HD + i * 16;
            uint4 v0 = *(const uint4*)(src);
            uint4 v1 = *(const uint4*)(src + 8);
            const unsigned int uu[8] = {v0.x, v0.y, v0.z, v0.w,
                                        v1.x, v1.y, v1.z, v1.w};
#pragma unroll
            for (int j = 0; j < 8; ++j) {
                const int c = i * 16 + 2 * j;
                sVt[(size_t)c * 40 + r]       = (unsigned short)(uu[j]);
                sVt[(size_t)(c + 1) * 40 + r] = (unsigned short)(uu[j] >> 16);
            }
        }
        __syncthreads();

        // S = Q @ K^T
        v8f s0 = zero8(), s1 = zero8();
#pragma unroll
        for (int kt = 0; kt < 2; ++kt) {
            v16bf bk0 = load_frag(sK, 72, 0,  kt * 32, lane);
            v16bf bk1 = load_frag(sK, 72, 16, kt * 32, lane);
            s0 = wmma_bf16(qF[kt], bk0, s0);
            s1 = wmma_bf16(qF[kt], bk1, s1);
        }

        // online softmax: row-max via 16-lane xor shuffles
        float alpha[8];
#pragma unroll
        for (int r = 0; r < 8; ++r) {
            const float a0 = s0[r] * 0.125f;   // 1/sqrt(64)
            const float a1 = s1[r] * 0.125f;
            float v = fmaxf(a0, a1);
            v = fmaxf(v, __shfl_xor(v, 1, 32));
            v = fmaxf(v, __shfl_xor(v, 2, 32));
            v = fmaxf(v, __shfl_xor(v, 4, 32));
            v = fmaxf(v, __shfl_xor(v, 8, 32));
            const float mn = fmaxf(m_[r], v);
            const float al = __expf(m_[r] - mn);
            m_[r] = mn;
            alpha[r] = al;
            unsigned short* pp = sP + (size_t)(w * 16 + r + hi) * 40;
            pp[(lane & 15)]      = f32_to_bf16(__expf(a0 - mn));
            pp[16 + (lane & 15)] = f32_to_bf16(__expf(a1 - mn));
        }
#pragma unroll
        for (int nt = 0; nt < 4; ++nt)
#pragma unroll
            for (int r = 0; r < 8; ++r) o[nt][r] *= alpha[r];

        __syncthreads();
        v16bf aP = load_frag(sP, 40, w * 16, 0, lane);
        // row sums of P via matrix pipe: P @ ones -> every column = row sum
        v8f rsum = wmma_bf16(aP, oneF, zero8());
#pragma unroll
        for (int r = 0; r < 8; ++r) l_[r] = l_[r] * alpha[r] + rsum[r];
        // O += P @ V
#pragma unroll
        for (int nt = 0; nt < 4; ++nt) {
            v16bf bv = load_frag(sVt, 40, nt * 16, 0, lane);
            o[nt] = wmma_bf16(aP, bv, o[nt]);
        }
    }

#pragma unroll
    for (int r = 0; r < 8; ++r) l_[r] = 1.0f / l_[r];
#pragma unroll
    for (int nt = 0; nt < 4; ++nt)
#pragma unroll
        for (int r = 0; r < 8; ++r) {
            const int grow = q0 + w * 16 + r + hi;
            const int gcol = nt * 16 + (lane & 15);
            Og[base + (size_t)grow * HD + gcol] = f32_to_bf16(o[nt][r] * l_[r]);
        }
}

// ---------------------------------------------------------------------------
// Kernel 5: residual add + LayerNorm, in place over d_out rows.
// ---------------------------------------------------------------------------
__global__ __launch_bounds__(256) void ln_kernel(
    float* __restrict__ out, const float* __restrict__ resid,
    const float* __restrict__ gamma, const float* __restrict__ beta) {
    const int row = blockIdx.x;
    const int t = threadIdx.x;
    __shared__ float rs[256], rq[256];
    float x[4];
    float s = 0.0f, sq = 0.0f;
#pragma unroll
    for (int i = 0; i < 4; ++i) {
        const int c = t + i * 256;
        const float v = out[(size_t)row * D_MODEL + c] +
                        resid[(size_t)row * D_MODEL + c];
        x[i] = v; s += v; sq += v * v;
    }
    rs[t] = s; rq[t] = sq;
    __syncthreads();
    for (int off = 128; off > 0; off >>= 1) {
        if (t < off) { rs[t] += rs[t + off]; rq[t] += rq[t + off]; }
        __syncthreads();
    }
    const float mu  = rs[0] * (1.0f / D_MODEL);
    const float var = rq[0] * (1.0f / D_MODEL) - mu * mu;
    const float inv = rsqrtf(var + 1e-5f);
#pragma unroll
    for (int i = 0; i < 4; ++i) {
        const int c = t + i * 256;
        out[(size_t)row * D_MODEL + c] =
            (x[i] - mu) * inv * gamma[c] + beta[c];
    }
}

// ---------------------------------------------------------------------------
// Host-side launch
// ---------------------------------------------------------------------------
extern "C" void kernel_launch(void* const* d_in, const int* in_sizes, int n_in,
                              void* d_out, int out_size, void* d_ws, size_t ws_size,
                              hipStream_t stream) {
    const float* preQ  = (const float*)d_in[0];
    const float* preK  = (const float*)d_in[1];
    const float* preV  = (const float*)d_in[2];
    const float* Wq    = (const float*)d_in[3];
    const float* Wk    = (const float*)d_in[4];
    const float* Wv    = (const float*)d_in[5];
    const float* Wo    = (const float*)d_in[6];
    const float* gamma = (const float*)d_in[7];
    const float* beta  = (const float*)d_in[8];
    float* out = (float*)d_out;

    char* ws = (char*)d_ws;
    constexpr size_t WT_BYTES  = (size_t)D_MODEL * D_MODEL * 2;   // 2 MB
    constexpr size_t XB_BYTES  = (size_t)M_ROWS * D_MODEL * 2;    // 16 MB
    unsigned short* WqT = (unsigned short*)(ws + 0 * WT_BYTES);
    unsigned short* WkT = (unsigned short*)(ws + 1 * WT_BYTES);
    unsigned short* WvT = (unsigned short*)(ws + 2 * WT_BYTES);
    unsigned short* WoT = (unsigned short*)(ws + 3 * WT_BYTES);
    char* xbase = ws + 4 * WT_BYTES;
    unsigned short* Xq = (unsigned short*)(xbase + 0 * XB_BYTES);
    unsigned short* Xk = (unsigned short*)(xbase + 1 * XB_BYTES);
    unsigned short* Xv = (unsigned short*)(xbase + 2 * XB_BYTES);
    unsigned short* Qb = (unsigned short*)(xbase + 3 * XB_BYTES);
    unsigned short* Kb = (unsigned short*)(xbase + 4 * XB_BYTES);
    unsigned short* Vb = (unsigned short*)(xbase + 5 * XB_BYTES);
    unsigned short* Ab = Xq;   // Xq dead after Q projection; reuse for attn out

    // 1) weights -> bf16 transposed [N][K]
    dim3 tb(32, 8), tg(D_MODEL / 32, D_MODEL / 32);
    wtrans_kernel<<<tg, tb, 0, stream>>>(Wq, WqT, D_MODEL);
    wtrans_kernel<<<tg, tb, 0, stream>>>(Wk, WkT, D_MODEL);
    wtrans_kernel<<<tg, tb, 0, stream>>>(Wv, WvT, D_MODEL);
    wtrans_kernel<<<tg, tb, 0, stream>>>(Wo, WoT, D_MODEL);

    // 2) activations -> bf16
    const int n8 = (M_ROWS * D_MODEL) / 8;       // 1M vec8 groups
    cvt_bf16_kernel<<<n8 / 256, 256, 0, stream>>>(preQ, Xq, n8);
    cvt_bf16_kernel<<<n8 / 256, 256, 0, stream>>>(preK, Xk, n8);
    cvt_bf16_kernel<<<n8 / 256, 256, 0, stream>>>(preV, Xv, n8);

    // 3) projections (bf16 in, bf16 out)
    dim3 gg(D_MODEL / 128, M_ROWS / 256);   // (8, 32)
    gemm_bf16_wmma_kernel<false><<<gg, 256, 0, stream>>>(
        Xq, WqT, Qb, M_ROWS, D_MODEL, D_MODEL);
    gemm_bf16_wmma_kernel<false><<<gg, 256, 0, stream>>>(
        Xk, WkT, Kb, M_ROWS, D_MODEL, D_MODEL);
    gemm_bf16_wmma_kernel<false><<<gg, 256, 0, stream>>>(
        Xv, WvT, Vb, M_ROWS, D_MODEL, D_MODEL);

    // 4) flash attention over 64 (b,h) slabs x 16 query blocks
    attn_kernel<<<dim3(S_LEN / 128, N_BATCH * N_HEADS), 256, 0, stream>>>(
        Qb, Kb, Vb, Ab);

    // 5) output projection straight into d_out (f32)
    gemm_bf16_wmma_kernel<true><<<gg, 256, 0, stream>>>(
        Ab, WoT, out, M_ROWS, D_MODEL, D_MODEL);

    // 6) residual + LayerNorm in place
    ln_kernel<<<M_ROWS, 256, 0, stream>>>(out, preV, gamma, beta);
}
